// QuantizedLinear_60722247631708
// MI455X (gfx1250) — 8-bit matrix operand order repaired and checked by reading, compiled, not run on hardware
//
#include <hip/hip_runtime.h>
#include <hip/hip_bf16.h>

typedef int   v8i  __attribute__((ext_vector_type(8)));
typedef int   v4i  __attribute__((ext_vector_type(4)));
typedef int   v2i  __attribute__((ext_vector_type(2)));
typedef float v4f  __attribute__((ext_vector_type(4)));

#define BM 128
#define BN 128
#define BKB 128
#define LDS_STRIDE 144
#define A_BYTES (BM * LDS_STRIDE)
#define B_BYTES (BN * LDS_STRIDE)

__global__ __launch_bounds__(256)
void quantize_x_kernel(const float* __restrict__ x,
                       const float* __restrict__ input_scale,
                       int* __restrict__ xq_packed, long long n4) {
    long long i = (long long)blockIdx.x * blockDim.x + threadIdx.x;
    if (i >= n4) return;
    const float inv = 1.0f / input_scale[0];
    v4f v = *(const v4f*)(x + i * 4);
    int r0 = (int)fminf(fmaxf(__builtin_rintf(v.x * inv), -128.0f), 127.0f);
    int r1 = (int)fminf(fmaxf(__builtin_rintf(v.y * inv), -128.0f), 127.0f);
    int r2 = (int)fminf(fmaxf(__builtin_rintf(v.z * inv), -128.0f), 127.0f);
    int r3 = (int)fminf(fmaxf(__builtin_rintf(v.w * inv), -128.0f), 127.0f);
    xq_packed[i] = (r0 & 255) | ((r1 & 255) << 8) | ((r2 & 255) << 16) | ((r3 & 255) << 24);
}

__global__ __launch_bounds__(256)
void pack_w_kernel(const int* __restrict__ w,
                   int* __restrict__ wq_packed, long long n4) {
    long long i = (long long)blockIdx.x * blockDim.x + threadIdx.x;
    if (i >= n4) return;
    v4i v = *(const v4i*)(w + i * 4);
    wq_packed[i] = (v.x & 255) | ((v.y & 255) << 8) | ((v.z & 255) << 16) | ((v.w & 255) << 24);
}

__global__ __launch_bounds__(256)
void gemm_i8_kernel(const char* __restrict__ aq,
                    const char* __restrict__ wq,
                    const float* __restrict__ wscale,
                    const float* __restrict__ iscale,
                    const int* __restrict__ bias_int,
                    const float* __restrict__ bscale,
                    float* __restrict__ out,
                    int M, int N, int K) {
    __shared__ int smem[(A_BYTES + B_BYTES) / 4];
    char* smA = (char*)smem;
    char* smB = smA + A_BYTES;

    const int tid    = threadIdx.x;
    const int lane   = tid & 31;
    const int wv     = tid >> 5;
    const int waveM  = wv >> 2;
    const int waveN  = wv & 3;
    const int lane16 = lane & 15;
    const int half   = lane >> 4;

    const int nBase = blockIdx.x * BN;
    const int mBase = blockIdx.y * BM;

    const v8i vzero = {0, 0, 0, 0, 0, 0, 0, 0};
    v8i acc[4][2];
#pragma unroll
    for (int i = 0; i < 4; ++i)
#pragma unroll
        for (int j = 0; j < 2; ++j) acc[i][j] = vzero;

    for (int kk = 0; kk < K; kk += BKB) {
        v4i aReg[4], bReg[4];
#pragma unroll
        for (int t = 0; t < 4; ++t) {
            int c = tid + t * 256;
            int row = c >> 3, col = c & 7;
            aReg[t] = *(const v4i*)(aq + (size_t)(mBase + row) * K + kk + col * 16);
            bReg[t] = *(const v4i*)(wq + (size_t)(nBase + row) * K + kk + col * 16);
        }
        if (kk + BKB < K) {
            __builtin_prefetch(aq + (size_t)(mBase + (tid >> 3)) * K + kk + BKB + (tid & 7) * 16, 0, 0);
            __builtin_prefetch(wq + (size_t)(nBase + (tid >> 3)) * K + kk + BKB + (tid & 7) * 16, 0, 0);
        }
        __syncthreads();
#pragma unroll
        for (int t = 0; t < 4; ++t) {
            int c = tid + t * 256;
            int row = c >> 3, col = c & 7;
            *(v4i*)(smA + row * LDS_STRIDE + col * 16) = aReg[t];
            *(v4i*)(smB + row * LDS_STRIDE + col * 16) = bReg[t];
        }
        __syncthreads();

#pragma unroll
        for (int ks = 0; ks < BKB; ks += 64) {
            v8i afrag[4];
#pragma unroll
            for (int mt = 0; mt < 4; ++mt) {
                const char* p = smA + (waveM * 64 + mt * 16 + lane16) * LDS_STRIDE + ks + half * 16;
                v2i t0 = *(const v2i*)(p);
                v2i t1 = *(const v2i*)(p + 8);
                v2i t2 = *(const v2i*)(p + 32);
                v2i t3 = *(const v2i*)(p + 40);
                v8i a;
                a[0] = t0.x; a[1] = t0.y; a[2] = t1.x; a[3] = t1.y;
                a[4] = t2.x; a[5] = t2.y; a[6] = t3.x; a[7] = t3.y;
                afrag[mt] = a;
            }
            v8i bfrag[2];
#pragma unroll
            for (int nt = 0; nt < 2; ++nt) {
                const char* p = smB + (waveN * 32 + nt * 16 + lane16) * LDS_STRIDE + ks + half * 16;
                v4i t0 = *(const v4i*)(p);
                v4i t1 = *(const v4i*)(p + 32);
                v8i b;
                b[0] = t0.x; b[1] = t0.y; b[2] = t0.z; b[3] = t0.w;
                b[4] = t1.x; b[5] = t1.y; b[6] = t1.z; b[7] = t1.w;
                bfrag[nt] = b;
            }
#pragma unroll
            for (int mt = 0; mt < 4; ++mt)
#pragma unroll
                for (int nt = 0; nt < 2; ++nt)
                    acc[mt][nt] = __builtin_amdgcn_wmma_i32_16x16x64_iu8(
                         true, afrag[mt],  true, bfrag[nt],
                        acc[mt][nt],  false,  false);
        }
    }

    const float is = iscale[0];
#pragma unroll
    for (int nt = 0; nt < 2; ++nt) {
        int n = nBase + waveN * 32 + nt * 16 + lane16;
        float sc = is * wscale[n];
        float bf = (float)bias_int[n] * bscale[n];
#pragma unroll
        for (int mt = 0; mt < 4; ++mt) {
            int mrow = mBase + waveM * 64 + mt * 16 + half * 8;
#pragma unroll
            for (int r = 0; r < 8; ++r) {
                out[(size_t)(mrow + r) * N + n] = (float)acc[mt][nt][r] * sc + bf;
            }
        }
    }
}

extern "C" void kernel_launch(void* const* d_in, const int* in_sizes, int n_in,
                              void* d_out, int out_size, void* d_ws, size_t ws_size,
                              hipStream_t stream) {
    const float* x     = (const float*)d_in[0];
    const int*   w     = (const int*)d_in[1];
    const float* wsc   = (const float*)d_in[2];
    const float* isc   = (const float*)d_in[4];
    const int*   bint  = (const int*)d_in[6];
    const float* bsc   = (const float*)d_in[7];
    float* out = (float*)d_out;

    const int N = in_sizes[2];
    const int K = in_sizes[1] / N;
    const int M = in_sizes[0] / K;

    char* aq = (char*)d_ws;
    char* wq = aq + (size_t)M * K;

    {
        long long n4 = (long long)M * K / 4;
        long long blocks = (n4 + 255) / 256;
        quantize_x_kernel<<<(unsigned)blocks, 256, 0, stream>>>(x, isc, (int*)aq, n4);
    }
    {
        long long n4 = (long long)N * K / 4;
        long long blocks = (n4 + 255) / 256;
        pack_w_kernel<<<(unsigned)blocks, 256, 0, stream>>>(w, (int*)wq, n4);
    }
    {
        dim3 grid(N / BN, M / BM);
        gemm_i8_kernel<<<grid, 256, 0, stream>>>(aq, wq, wsc, isc, bint, bsc, out, M, N, K);
    }
}
